// DetectorLoss_15985868275745
// MI455X (gfx1250) — compile-verified, matrix operand order checked by reading, not run on hardware
//
#include <hip/hip_runtime.h>

// ---------------- problem constants (match reference) ----------------
#define BDIM   32
#define ADIM   3
#define CDIM   20
#define HDIM   160
#define WDIM   160
#define HWDIM  (HDIM * WDIM)               // 25600
#define NCELLS (BDIM * ADIM * HWDIM)       // 2,457,600
#define NCELLS4 (NCELLS / 4)               // 614,400 float4s

// Dedup hash table: 65536 entries (256 KB) >> max possible P (24000).
#define HASH_SIZE 65536

typedef __attribute__((ext_vector_type(2)))  float    v2f;
typedef __attribute__((ext_vector_type(8)))  float    v8f;
typedef __attribute__((ext_vector_type(16))) _Float16 v16h;

#if defined(__has_builtin)
#  if __has_builtin(__builtin_amdgcn_wmma_f32_16x16x4_f32)
#    define HAVE_WMMA_F32X4 1
#  endif
#endif

// Wave32 cross-lane sum via V_WMMA_F32_16X16X4_F32 with all-ones B:
//   D[m][n] = sum_k A[m][k] -> row m = v[lane m] + v[lane m+16]
// Lane 0's 8 C/D VGPRs cover rows 0..7, lane 16's cover rows 8..15, so
// lane0+lane16 = full wave sum (caller accumulates from lanes (tid&15)==0).
// Full f32 precision; EXEC is all-ones at every call site (no early returns).
__device__ __forceinline__ float wmma_wave_sum(float v) {
#ifdef HAVE_WMMA_F32X4
  v2f a; a[0] = v;     a[1] = 0.0f;
  v2f b; b[0] = 1.0f;  b[1] = 1.0f;
  v8f c = {};
  c = __builtin_amdgcn_wmma_f32_16x16x4_f32(false, a, false, b, (short)0, c,
                                            false, false);
#else
  v16h a = {};
  a[0] = (_Float16)v;
  v16h b;
#pragma unroll
  for (int i = 0; i < 16; ++i) b[i] = (_Float16)1.0f;
  v8f c = {};
  c = __builtin_amdgcn_wmma_f32_16x16x32_f16(false, a, false, b, (short)0, c,
                                             false, false);
#endif
  return c[0] + c[1] + c[2] + c[3] + c[4] + c[5] + c[6] + c[7];
}

__device__ __forceinline__ float smooth_l1(float d) {
  float ad = fabsf(d);
  return (ad < 1.0f) ? 0.5f * d * d : (ad - 0.5f);
}

// SIoU (box format: cx, cy, w, h). angle_cost simplified analytically:
// cos(2*asin(s) - pi/2) == 2*s*sqrt(1 - s^2).
__device__ __forceinline__ float siou_fn(float p_x, float p_y, float p_w, float p_h,
                                         float g_x, float g_y, float g_w, float g_h) {
  const float eps = 1e-7f;
  float b1x1 = p_x - p_w * 0.5f, b1x2 = p_x + p_w * 0.5f;
  float b1y1 = p_y - p_h * 0.5f, b1y2 = p_y + p_h * 0.5f;
  float b2x1 = g_x - g_w * 0.5f, b2x2 = g_x + g_w * 0.5f;
  float b2y1 = g_y - g_h * 0.5f, b2y2 = g_y + g_h * 0.5f;

  float iw = fminf(b1x2, b2x2) - fmaxf(b1x1, b2x1);
  float ih = fminf(b1y2, b2y2) - fmaxf(b1y1, b2y1);
  float inter = fmaxf(iw, 0.0f) * fmaxf(ih, 0.0f);

  float w1 = b1x2 - b1x1, h1 = b1y2 - b1y1 + eps;
  float w2 = b2x2 - b2x1, h2 = b2y2 - b2y1 + eps;
  float uni = w1 * h1 + w2 * h2 - inter + eps;
  float iou = inter / uni;

  float cw = fmaxf(b1x2, b2x2) - fminf(b1x1, b2x1);
  float ch = fmaxf(b1y2, b2y2) - fminf(b1y1, b2y1);
  float s_cw = (b2x1 + b2x2 - b1x1 - b1x2) * 0.5f;
  float s_ch = (b2y1 + b2y2 - b1y1 - b1y2) * 0.5f;
  float sigma = sqrtf(s_cw * s_cw + s_ch * s_ch);
  float sin_a1 = fabsf(s_cw) / sigma;
  float sin_a2 = fabsf(s_ch) / sigma;
  const float thr = 0.70710678118654752f;
  float sa = (sin_a1 > thr) ? sin_a2 : sin_a1;
  float angle_cost = 2.0f * sa * sqrtf(fmaxf(1.0f - sa * sa, 0.0f));

  float rx = s_cw / cw; rx *= rx;
  float ry = s_ch / ch; ry *= ry;
  float gamma = angle_cost - 2.0f;
  float dist_cost = 2.0f - __expf(gamma * rx) - __expf(gamma * ry);

  float ow = fabsf(w1 - w2) / fmaxf(w1, w2);
  float oh = fabsf(h1 - h2) / fmaxf(h1, h2);
  float e1 = 1.0f - __expf(-ow);
  float e2 = 1.0f - __expf(-oh);
  float shape_cost = e1 * e1 * e1 * e1 + e2 * e2 * e2 * e2;

  return iou - 0.5f * (dist_cost + shape_cost);
}

// ---------------- kernels ----------------

// Zero accumulators + per-batch counts; set hash table to -1 via B128 stores.
// Launch: 64 blocks x 256 threads -> 16384 int4 = 65536 hash entries exactly.
__global__ void k_init(float* __restrict__ accF, int* __restrict__ n_b,
                       int4* __restrict__ hash4, int use_hash) {
  int idx = blockIdx.x * blockDim.x + threadIdx.x;
  if (idx < 4) accF[idx] = 0.0f;
  if (idx < BDIM) n_b[idx] = 0;
  if (use_hash && idx < (HASH_SIZE / 4)) {
    hash4[idx] = make_int4(-1, -1, -1, -1);
  }
}

// Per-positive: SIoU vs gt box, cls NLL, bincount(p_b). WMMA wave reduce ->
// LDS ds_add_f32 -> one global atomic per block.
__global__ void k_pos1(const float* __restrict__ pdb, const float* __restrict__ pcls,
                       const float* __restrict__ gt_box, const int* __restrict__ gt_cls,
                       const int* __restrict__ pb, const int* __restrict__ pxi,
                       const int* __restrict__ pyi, const int* __restrict__ pai,
                       const float* __restrict__ anchors,
                       float* __restrict__ iou_vals, int* __restrict__ n_b,
                       float* __restrict__ accF, int P) {
  __shared__ float s_iou, s_cls;
  if (threadIdx.x == 0) { s_iou = 0.0f; s_cls = 0.0f; }
  __syncthreads();

  int p = blockIdx.x * blockDim.x + threadIdx.x;
  int active = (p < P);
  int q = active ? p : 0;

  int b = pb[q], a = pai[q], x = pxi[q], y = pyi[q];
  int base = ((b * ADIM + a) * 4) * HWDIM + y * WDIM + x;
  float d0 = pdb[base];
  float d1 = pdb[base + HWDIM];
  float d2 = pdb[base + 2 * HWDIM];
  float d3 = pdb[base + 3 * HWDIM];

  float aw = anchors[2 * a] * (float)WDIM;
  float ah = anchors[2 * a + 1] * (float)HDIM;
  float pw = __expf(d2) * aw;
  float ph = __expf(d3) * ah;
  float pxc = tanhf(d0) + (float)x;
  float pyc = tanhf(d1) + (float)y;

  float gx = gt_box[4 * q + 0], gy = gt_box[4 * q + 1];
  float gw = gt_box[4 * q + 2], gh = gt_box[4 * q + 3];
  float iou = siou_fn(pxc, pyc, pw, ph, gx, gy, gw, gh);

  int ci = (((b * ADIM + a) * CDIM + gt_cls[q]) * HDIM + y) * WDIM + x;
  float lcls = -__logf(pcls[ci]);

  if (active) {
    iou_vals[p] = iou;
    atomicAdd(&n_b[b], 1);
  }
  float v_iou = active ? (1.0f - iou) : 0.0f;
  float v_cls = active ? lcls : 0.0f;

  float r0 = wmma_wave_sum(v_iou);
  float r1 = wmma_wave_sum(v_cls);
  if ((threadIdx.x & 15u) == 0u) { atomicAdd(&s_iou, r0); atomicAdd(&s_cls, r1); }
  __syncthreads();
  if (threadIdx.x == 0) { atomicAdd(&accF[0], s_iou); atomicAdd(&accF[1], s_cls); }
}

// Per-positive correction to obj loss:
//   corr = sl1(po - iou) * (0.25*H*W/n[b]) - 0.75 * sl1(po)
// Open-addressing hash (atomicCAS, linear probing) picks exactly one winner
// per duplicated cell, matching .at[].set single-winner semantics, without a
// dense 9.8 MB claim map.
__global__ void k_pos2(const float* __restrict__ pobj, const float* __restrict__ iou_vals,
                       const int* __restrict__ n_b,
                       const int* __restrict__ pb, const int* __restrict__ pxi,
                       const int* __restrict__ pyi, const int* __restrict__ pai,
                       int* __restrict__ hash, float* __restrict__ accF,
                       int P, int use_hash) {
  __shared__ float s_corr;
  if (threadIdx.x == 0) s_corr = 0.0f;
  __syncthreads();

  int p = blockIdx.x * blockDim.x + threadIdx.x;
  int active = (p < P);
  int q = active ? p : 0;

  int b = pb[q], a = pai[q], x = pxi[q], y = pyi[q];
  int cell = ((b * ADIM + a) * HDIM + y) * WDIM + x;

  int won = active;
  if (active && use_hash) {
    unsigned h = ((unsigned)cell * 2654435761u) >> 16;  // Knuth multiplicative
    h &= (HASH_SIZE - 1);
    won = 0;
    for (int t = 0; t < HASH_SIZE; ++t) {
      int old = atomicCAS(&hash[h], -1, cell);
      if (old == -1) { won = 1; break; }   // claimed this cell
      if (old == cell) { won = 0; break; } // another positive owns it
      h = (h + 1) & (HASH_SIZE - 1);
    }
  }

  float po = pobj[cell];
  float iou = iou_vals[q];
  int n = n_b[b]; if (n < 1) n = 1;
  float fac = 0.25f * (float)HWDIM / (float)n;
  float corr = smooth_l1(po - iou) * fac - 0.75f * smooth_l1(po);
  float v = won ? corr : 0.0f;

  float r = wmma_wave_sum(v);
  if ((threadIdx.x & 15u) == 0u) atomicAdd(&s_corr, r);
  __syncthreads();
  if (threadIdx.x == 0) atomicAdd(&accF[3], s_corr);
}

// Dense base pass: sum 0.75 * smoothL1(pred_obj) over all cells (float4 loads).
// 600 blocks * 256 threads -> exactly 4 iterations per thread, no tail
// divergence, so EXEC is all-ones at the WMMA reduce.
__global__ void k_obj(const float4* __restrict__ po4, float* __restrict__ accF) {
  __shared__ float s_obj;
  if (threadIdx.x == 0) s_obj = 0.0f;
  __syncthreads();

  int tid = blockIdx.x * blockDim.x + threadIdx.x;
  int stride = gridDim.x * blockDim.x;
  float acc = 0.0f;
  for (int i = tid; i < NCELLS4; i += stride) {
    float4 v = po4[i];
    __builtin_prefetch(po4 + i + stride, 0, 1);  // global_prefetch_b8
    acc += smooth_l1(v.x) + smooth_l1(v.y) + smooth_l1(v.z) + smooth_l1(v.w);
  }
  acc *= 0.75f;

  float r = wmma_wave_sum(acc);
  if ((threadIdx.x & 15u) == 0u) atomicAdd(&s_obj, r);
  __syncthreads();
  if (threadIdx.x == 0) atomicAdd(&accF[2], s_obj);
}

__global__ void k_fin(const float* __restrict__ accF, float* __restrict__ out, int P) {
  float invP = 1.0f / (float)(P > 0 ? P : 1);
  float iou_loss = accF[0] * invP;
  float cls_loss = accF[1] * invP;
  float obj_loss = (accF[2] + accF[3]) * (1.0f / (float)NCELLS);
  out[0] = iou_loss;
  out[1] = obj_loss;
  out[2] = cls_loss;
  out[3] = iou_loss + 4.0f * obj_loss + 2.0f * cls_loss;
}

// ---------------- host launcher ----------------
extern "C" void kernel_launch(void* const* d_in, const int* in_sizes, int n_in,
                              void* d_out, int out_size, void* d_ws, size_t ws_size,
                              hipStream_t stream) {
  const float* pred_obj = (const float*)d_in[0];
  const float* pdb      = (const float*)d_in[1];
  const float* pcls     = (const float*)d_in[2];
  const float* gt_box   = (const float*)d_in[3];
  const int*   gt_cls   = (const int*)d_in[4];
  const int*   p_b      = (const int*)d_in[5];
  const int*   p_x      = (const int*)d_in[6];
  const int*   p_y      = (const int*)d_in[7];
  const int*   p_a      = (const int*)d_in[8];
  const float* anchors  = (const float*)d_in[9];

  const int P = in_sizes[4];  // gt_cls count == number of positives

  // Workspace: [accF(4f) pad64][n_b(32i) pad to 256][iou_vals(P f) pad256][hash 256KB]
  char*  ws       = (char*)d_ws;
  float* accF     = (float*)ws;
  int*   n_b      = (int*)(ws + 64);
  float* iou_vals = (float*)(ws + 256);
  size_t iou_bytes = (((size_t)(P > 0 ? P : 1) * 4) + 255) & ~(size_t)255;
  int*   hash     = (int*)(ws + 256 + iou_bytes);
  size_t need     = 256 + iou_bytes + (size_t)HASH_SIZE * 4;
  int use_hash    = (ws_size >= need) ? 1 : 0;

  const int TPB = 256;
  int init_blocks = use_hash ? (HASH_SIZE / 4 / TPB) : 1;  // 64 blocks
  k_init<<<init_blocks, TPB, 0, stream>>>(accF, n_b, (int4*)hash, use_hash);

  if (P > 0) {
    int pos_blocks = (P + TPB - 1) / TPB;
    k_pos1<<<pos_blocks, TPB, 0, stream>>>(pdb, pcls, gt_box, gt_cls,
                                           p_b, p_x, p_y, p_a, anchors,
                                           iou_vals, n_b, accF, P);
    k_pos2<<<pos_blocks, TPB, 0, stream>>>(pred_obj, iou_vals, n_b,
                                           p_b, p_x, p_y, p_a,
                                           hash, accF, P, use_hash);
  }

  k_obj<<<600, TPB, 0, stream>>>((const float4*)pred_obj, accF);

  k_fin<<<1, 1, 0, stream>>>(accF, (float*)d_out, P);
}
